// GlobalViewPathScorerGNN_80796924772858
// MI455X (gfx1250) — matrix operand order repaired and checked by reading, not compiled, not run on hardware
//
#include <hip/hip_runtime.h>
#include <hip/hip_bf16.h>

typedef __bf16 bf16_t;
typedef __attribute__((ext_vector_type(16))) __bf16 v16bf;
typedef __attribute__((ext_vector_type(8)))  float  v8f;
typedef unsigned int v4u __attribute__((ext_vector_type(4)));
typedef int          v8i __attribute__((ext_vector_type(8)));
typedef int          v4i __attribute__((ext_vector_type(4)));

constexpr int N_NODES = 131072;
constexpr int F_NODE  = 64;
constexpr int F_FLOW  = 16;
constexpr int H_DIM   = 128;
constexpr int N_EDGES = 2097152;
constexpr int BATCH   = 256;
constexpr int L_PATH  = 32;

// LSTM dynamic LDS: h double-buffer + both weight matrices (bf16)
constexpr size_t LSTM_SMEM_ELEMS = (size_t)2 * 16 * H_DIM + 2 * (size_t)4 * H_DIM * H_DIM;
constexpr size_t LSTM_SMEM_BYTES = LSTM_SMEM_ELEMS * sizeof(bf16_t);   // 264 KB < 320 KB

// ---------------------------------------------------------------------------
// WMMA fragment loaders (layouts per CDNA5 ISA 7.12.2, wave32)
// A: 16x32 bf16. lane&15 = M row. lanes 0-15: elems 0..7 = K+0..7, 8..15 = K+16..23
//                               lanes 16-31: elems 0..7 = K+8..15, 8..15 = K+24..31
__device__ __forceinline__ v16bf load_a_rowptr(const bf16_t* rowp, int kk, int lane) {
  const bf16_t* p = rowp + kk + ((lane >> 4) << 3);
  v16bf a;
#pragma unroll
  for (int j = 0; j < 8; ++j) { a[j] = p[j]; a[j + 8] = p[j + 16]; }
  return a;
}

// B: 32x16 bf16 with B[k][n] = W[n][k], W row-major (rows = out features, ld = K).
// lane&15 = N column; lanes 0-15: elems 0..7 = K+0..7, 8..15 = K+16..23; lanes 16-31: K+8..15, K+24..31.
__device__ __forceinline__ v16bf load_b_wt(const bf16_t* W, int ldw, int n0, int kk, int lane) {
  const bf16_t* p = W + (size_t)(n0 + (lane & 15)) * ldw + kk + ((lane >> 4) << 3);
  v16bf b;
#pragma unroll
  for (int j = 0; j < 8; ++j) { b[j] = p[j]; b[j + 8] = p[j + 16]; }
  return b;
}

__device__ __forceinline__ float sigmf(float x) { return 1.0f / (1.0f + __expf(-x)); }

// ---------------------------------------------------------------------------
// Tensor Data Mover: stage a 2D row-major bf16 matrix (dim1 rows x dim0 cols)
// from global memory into LDS. D# layout per CDNA5 ISA ch.8.
// group0: [1:0]=count=1 | lds_addr | global_addr[56:0] | type=2
// group1: data_size=1 (2B), tensor_dim0/1, tile_dim0/1 (full matrix), stride0=dim0
// 6-arg builtin: (uint32x4 g0, int32x8 g1, int32x4 g2, int32x4 g3, int32x8, i32 cpol)
__device__ __forceinline__ void tdm_load_2d_bf16(unsigned int lds_off, const void* gptr,
                                                 unsigned int dim0, unsigned int dim1) {
  unsigned long long ga = (unsigned long long)(uintptr_t)gptr;
  v4u g0;
  g0[0] = 1u;                                     // count=1 (valid user D#)
  g0[1] = lds_off;                                // lds_addr (bytes)
  g0[2] = (unsigned int)ga;                       // global_addr[31:0]
  g0[3] = (unsigned int)(ga >> 32) | (2u << 30);  // global_addr[56:32] | type=2
  v8i g1;
  g1[0] = (int)(1u << 16);                                   // data_size=1 (2 bytes)
  g1[1] = (int)((dim0 & 0xFFFFu) << 16);                     // tensor_dim0[15:0] @63:48
  g1[2] = (int)((dim0 >> 16) | ((dim1 & 0xFFFFu) << 16));    // t_dim0[31:16], t_dim1[15:0]
  g1[3] = (int)((dim1 >> 16) | ((dim0 & 0xFFFFu) << 16));    // t_dim1[31:16], tile_dim0
  g1[4] = (int)(dim1 & 0xFFFFu);                             // tile_dim1 (tile_dim2=0)
  g1[5] = (int)dim0;                                         // tensor_dim0_stride[31:0]
  g1[6] = 0;
  g1[7] = 0;
  v4i z4 = {0, 0, 0, 0};
  v8i z8 = {0, 0, 0, 0, 0, 0, 0, 0};
  __builtin_amdgcn_tensor_load_to_lds(g0, g1, z4, z4, z8, 0);
}

// ---------------------------------------------------------------------------
// Utility kernels
__global__ void zero_f32_kernel(float* __restrict__ p, size_t n) {
  size_t i = (size_t)blockIdx.x * blockDim.x + threadIdx.x;
  if (i < n) p[i] = 0.0f;
}

__global__ void f32_to_bf16_kernel(const float* __restrict__ s, bf16_t* __restrict__ d, size_t n) {
  size_t i = (size_t)blockIdx.x * blockDim.x + threadIdx.x;
  if (i < n) d[i] = (bf16_t)s[i];
}

// ---------------------------------------------------------------------------
// Edge scatter (segment_sum over dst) + degree count
__global__ void edge_count_kernel(const int* __restrict__ dst, float* __restrict__ cnt, int e) {
  int i = blockIdx.x * blockDim.x + threadIdx.x;
  if (i < e) atomicAdd(&cnt[dst[i]], 1.0f);
}

// one thread handles 4 consecutive features of one edge (float4 read, 4 atomics)
__global__ void edge_scatter4_kernel(const int* __restrict__ src, const int* __restrict__ dst,
                                     const float* __restrict__ feat, float* __restrict__ agg,
                                     int F, size_t totalq) {
  size_t tid = (size_t)blockIdx.x * blockDim.x + threadIdx.x;
  if (tid >= totalq) return;
  int Fq = F >> 2;
  size_t e = tid / Fq;
  int f = (int)(tid - e * Fq) << 2;
  int s = src[e], d = dst[e];
  const float4 v = *(const float4*)(feat + (size_t)s * F + f);
  float* base = agg + (size_t)d * F + f;
  atomicAdd(base + 0, v.x);
  atomicAdd(base + 1, v.y);
  atomicAdd(base + 2, v.z);
  atomicAdd(base + 3, v.w);
}

// mean = agg / max(cnt,1), emitted directly as bf16 for the WMMA GEMM
__global__ void finalize_mean_kernel(const float* __restrict__ agg, const float* __restrict__ cnt,
                                     bf16_t* __restrict__ outb, int F, size_t total) {
  size_t tid = (size_t)blockIdx.x * blockDim.x + threadIdx.x;
  if (tid >= total) return;
  size_t node = tid / F;
  outb[tid] = (bf16_t)(agg[tid] / fmaxf(cnt[node], 1.0f));
}

// ---------------------------------------------------------------------------
// Fused SAGE layer: out = relu(mean @ Wl^T + self @ Wr^T + b).
// Weights TDM-staged into LDS once per block; each wave computes a full
// 16(M) x 128(N) row stripe with 8 f32 accumulators; A loaded once per K-step.
// All 8 B fragments loaded into registers before the WMMA burst so the
// ds_loads pipeline and the WMMAs issue back-to-back.
__global__ __launch_bounds__(256) void sage_gemm_kernel(
    const bf16_t* __restrict__ meanb, const bf16_t* __restrict__ selfb, int K,
    const bf16_t* __restrict__ Wlb, const bf16_t* __restrict__ Wrb,
    const float* __restrict__ bias, float* __restrict__ outf, bf16_t* __restrict__ outb) {
  __shared__ bf16_t wl_s[H_DIM * 128];   // 32 KB (K<=128)
  __shared__ bf16_t wr_s[H_DIM * 128];   // 32 KB
  const int lane = threadIdx.x & 31;
  const int wave = threadIdx.x >> 5;

  if (wave == 0) {   // one wave drives the two TDM transfers for the block
    tdm_load_2d_bf16((unsigned int)(uintptr_t)(void*)wl_s, Wlb, (unsigned)K, (unsigned)H_DIM);
    tdm_load_2d_bf16((unsigned int)(uintptr_t)(void*)wr_s, Wrb, (unsigned)K, (unsigned)H_DIM);
    __builtin_amdgcn_s_wait_tensorcnt(0);
  }
  __syncthreads();

  const int m    = blockIdx.x * 8 + wave;   // 16-row M tile per wave
  const int row0 = m << 4;
  const int rowA = row0 + (lane & 15);

  v8f c[8] = {};   // 8 x 16-wide N tiles

  const bf16_t* arow = meanb + (size_t)rowA * K;
  __builtin_prefetch(selfb + (size_t)rowA * K, 0, 1);   // global_prefetch
  for (int kk = 0; kk < K; kk += 32) {
    v16bf a = load_a_rowptr(arow, kk, lane);
    v16bf bfr[8];
#pragma unroll
    for (int n = 0; n < 8; ++n) bfr[n] = load_b_wt(wl_s, K, n << 4, kk, lane);
#pragma unroll
    for (int n = 0; n < 8; ++n)
      c[n] = __builtin_amdgcn_wmma_f32_16x16x32_bf16(false, a, false, bfr[n], (short)0, c[n], false, false);
  }
  arow = selfb + (size_t)rowA * K;
  for (int kk = 0; kk < K; kk += 32) {
    v16bf a = load_a_rowptr(arow, kk, lane);
    v16bf bfr[8];
#pragma unroll
    for (int n = 0; n < 8; ++n) bfr[n] = load_b_wt(wr_s, K, n << 4, kk, lane);
#pragma unroll
    for (int n = 0; n < 8; ++n)
      c[n] = __builtin_amdgcn_wmma_f32_16x16x32_bf16(false, a, false, bfr[n], (short)0, c[n], false, false);
  }

  // C/D layout: VGPR v -> M = v + 8*(lane>=16), N = lane&15
  const int mb = row0 + ((lane >> 4) << 3);
#pragma unroll
  for (int n = 0; n < 8; ++n) {
    const int nn = (n << 4) + (lane & 15);
    const float bj = bias[nn];
#pragma unroll
    for (int v = 0; v < 8; ++v) {
      float val = fmaxf(c[n][v] + bj, 0.0f);
      size_t idx = (size_t)(mb + v) * H_DIM + nn;
      outf[idx] = val;
      outb[idx] = (bf16_t)val;
    }
  }
}

// ---------------------------------------------------------------------------
// Graph mean pooling
__global__ void pool_sum_kernel(const float* __restrict__ emb, const int* __restrict__ batch,
                                float* __restrict__ gsum, size_t total) {
  size_t tid = (size_t)blockIdx.x * blockDim.x + threadIdx.x;
  if (tid >= total) return;
  size_t node = tid >> 7;
  int h = (int)(tid & 127);
  atomicAdd(&gsum[(size_t)batch[node] * H_DIM + h], emb[tid]);
}

__global__ void pool_cnt_kernel(const int* __restrict__ batch, float* __restrict__ gcnt, int n) {
  int i = blockIdx.x * blockDim.x + threadIdx.x;
  if (i < n) atomicAdd(&gcnt[batch[i]], 1.0f);
}

__global__ void pool_fin_kernel(const float* __restrict__ gsum, const float* __restrict__ gcnt,
                                float* __restrict__ gemb) {
  int tid = blockIdx.x * blockDim.x + threadIdx.x;   // BATCH*H
  if (tid >= BATCH * H_DIM) return;
  gemb[tid] = gsum[tid] / fmaxf(gcnt[tid >> 7], 1.0f);
}

// ---------------------------------------------------------------------------
// flow_emb = relu(flow @ Wf^T + bf)
__global__ void flow_kernel(const float* __restrict__ flow, const float* __restrict__ Wf,
                            const float* __restrict__ bf_, float* __restrict__ femb) {
  int b = blockIdx.x, j = threadIdx.x;   // 256 x 64
  float acc = bf_[j];
#pragma unroll
  for (int k = 0; k < F_FLOW; ++k) acc += flow[b * F_FLOW + k] * Wf[j * F_FLOW + k];
  femb[b * 64 + j] = fmaxf(acc, 0.0f);
}

// ---------------------------------------------------------------------------
// LSTM over path sequences. Batch tiles of 16 rows are independent:
// grid = 16 blocks, block = 256 threads = 8 waves (one 16-wide H column tile each).
// Wih/Whh TDM-staged into LDS once per block (reused 32 steps); h double-buffered
// in LDS; c kept in registers per lane.
__global__ __launch_bounds__(256) void lstm_kernel(
    const bf16_t* __restrict__ embb, const int* __restrict__ path_idx,
    const int* __restrict__ path_len,
    const bf16_t* __restrict__ Wihb, const bf16_t* __restrict__ Whhb,
    const float* __restrict__ bih, const float* __restrict__ bhh,
    float* __restrict__ pemb) {
  extern __shared__ bf16_t smem[];
  bf16_t* hbuf  = smem;                              // 2 * 16*H bf16
  bf16_t* wih_s = smem + (size_t)2 * 16 * H_DIM;     // 4H x H bf16
  bf16_t* whh_s = wih_s + (size_t)4 * H_DIM * H_DIM; // 4H x H bf16

  const int lane = threadIdx.x & 31;
  const int nh   = threadIdx.x >> 5;     // wave id = H column tile (0..7)
  const int m    = blockIdx.x;           // batch tile (0..15)
  const int row0 = m << 4;
  const int col0 = nh << 4;
  const int rA   = lane & 15;            // local A-row owned by this lane
  const int bbA  = row0 + rA;

  if (nh == 0) {   // one wave drives the weight staging for the block
    tdm_load_2d_bf16((unsigned int)(uintptr_t)(void*)wih_s, Wihb,
                     (unsigned)H_DIM, (unsigned)(4 * H_DIM));
    tdm_load_2d_bf16((unsigned int)(uintptr_t)(void*)whh_s, Whhb,
                     (unsigned)H_DIM, (unsigned)(4 * H_DIM));
    __builtin_amdgcn_s_wait_tensorcnt(0);
  }
  for (int i = threadIdx.x; i < 2 * 16 * H_DIM; i += 256) hbuf[i] = (bf16_t)0.0f;
  __syncthreads();

  float creg[8];
#pragma unroll
  for (int v = 0; v < 8; ++v) creg[v] = 0.0f;

  const int hh  = col0 + (lane & 15);     // owned output column
  const int mbl = (lane >> 4) << 3;       // owned local row base (0 or 8)
  const float bi0 = bih[hh]             + bhh[hh];
  const float bi1 = bih[H_DIM + hh]     + bhh[H_DIM + hh];
  const float bi2 = bih[2 * H_DIM + hh] + bhh[2 * H_DIM + hh];
  const float bi3 = bih[3 * H_DIM + hh] + bhh[3 * H_DIM + hh];

  for (int t = 0; t < L_PATH; ++t) {
    const bf16_t* hread  = hbuf + (t & 1) * (16 * H_DIM);
    bf16_t*       hwrite = hbuf + ((t + 1) & 1) * (16 * H_DIM);

    v8f acc[4] = {};   // i, f, g, o accumulators

    // x_t part: gather node row via path_idx
    const int nodeRow = path_idx[bbA * L_PATH + t];
    const bf16_t* xrow = embb + (size_t)nodeRow * H_DIM;
#pragma unroll
    for (int kk = 0; kk < H_DIM; kk += 32) {
      v16bf a = load_a_rowptr(xrow, kk, lane);
      v16bf bfr[4];
#pragma unroll
      for (int g = 0; g < 4; ++g) bfr[g] = load_b_wt(wih_s, H_DIM, g * H_DIM + col0, kk, lane);
#pragma unroll
      for (int g = 0; g < 4; ++g)
        acc[g] = __builtin_amdgcn_wmma_f32_16x16x32_bf16(false, a, false, bfr[g], (short)0, acc[g], false, false);
    }
    // h_{t-1} part from LDS
    const bf16_t* hrow = hread + rA * H_DIM;
#pragma unroll
    for (int kk = 0; kk < H_DIM; kk += 32) {
      v16bf a = load_a_rowptr(hrow, kk, lane);
      v16bf bfr[4];
#pragma unroll
      for (int g = 0; g < 4; ++g) bfr[g] = load_b_wt(whh_s, H_DIM, g * H_DIM + col0, kk, lane);
#pragma unroll
      for (int g = 0; g < 4; ++g)
        acc[g] = __builtin_amdgcn_wmma_f32_16x16x32_bf16(false, a, false, bfr[g], (short)0, acc[g], false, false);
    }

    // elementwise gate math on owned C elements
#pragma unroll
    for (int v = 0; v < 8; ++v) {
      const int rloc = mbl + v;
      const int bb   = row0 + rloc;
      float gi = acc[0][v] + bi0;
      float gf = acc[1][v] + bi1;
      float gg = acc[2][v] + bi2;
      float go = acc[3][v] + bi3;
      float cn = sigmf(gf) * creg[v] + sigmf(gi) * tanhf(gg);
      float hn = sigmf(go) * tanhf(cn);
      bool valid = t < path_len[bb];
      float hold = (float)hread[rloc * H_DIM + hh];
      if (valid) creg[v] = cn;
      hwrite[rloc * H_DIM + hh] = (bf16_t)(valid ? hn : hold);
    }
    __syncthreads();   // all of step t (reads of hread, writes of hwrite) done
  }

  const bf16_t* hf = hbuf + (L_PATH & 1) * (16 * H_DIM);
#pragma unroll
  for (int v = 0; v < 8; ++v) {
    const int rloc = mbl + v;
    pemb[(size_t)(row0 + rloc) * H_DIM + hh] = (float)hf[rloc * H_DIM + hh];
  }
}

// ---------------------------------------------------------------------------
// scores = relu([gemb|pemb|femb] @ Ws1^T + bs1) @ Ws2^T + bs2
__global__ __launch_bounds__(128) void score_kernel(
    const float* __restrict__ gemb, const float* __restrict__ pemb,
    const float* __restrict__ femb, const float* __restrict__ Ws1,
    const float* __restrict__ bs1, const float* __restrict__ Ws2,
    const float* __restrict__ bs2, float* __restrict__ out) {
  const int b = blockIdx.x, j = threadIdx.x;   // 256 x 128
  const float* w = Ws1 + j * 320;
  float acc = bs1[j];
  for (int k = 0; k < H_DIM; ++k) acc += gemb[b * H_DIM + k] * w[k];
  for (int k = 0; k < H_DIM; ++k) acc += pemb[b * H_DIM + k] * w[H_DIM + k];
  for (int k = 0; k < 64; ++k)    acc += femb[b * 64 + k]    * w[2 * H_DIM + k];
  float h = fmaxf(acc, 0.0f) * Ws2[j];
  __shared__ float red[128];
  red[j] = h;
  __syncthreads();
  for (int s = 64; s > 0; s >>= 1) {
    if (j < s) red[j] += red[j + s];
    __syncthreads();
  }
  if (j == 0) out[b] = red[0] + bs2[0];
}

// ---------------------------------------------------------------------------
extern "C" void kernel_launch(void* const* d_in, const int* in_sizes, int n_in,
                              void* d_out, int out_size, void* d_ws, size_t ws_size,
                              hipStream_t stream) {
  const float* x        = (const float*)d_in[0];
  const int*   eidx     = (const int*)d_in[1];
  const int*   src      = eidx;
  const int*   dst      = eidx + N_EDGES;
  const int*   batch    = (const int*)d_in[2];
  const int*   path_idx = (const int*)d_in[3];
  const int*   path_len = (const int*)d_in[4];
  const float* flow     = (const float*)d_in[5];
  const float* Wl1 = (const float*)d_in[6];
  const float* Wr1 = (const float*)d_in[7];
  const float* b1  = (const float*)d_in[8];
  const float* Wl2 = (const float*)d_in[9];
  const float* Wr2 = (const float*)d_in[10];
  const float* b2  = (const float*)d_in[11];
  const float* Wih = (const float*)d_in[12];
  const float* Whh = (const float*)d_in[13];
  const float* bih = (const float*)d_in[14];
  const float* bhh = (const float*)d_in[15];
  const float* Wf  = (const float*)d_in[16];
  const float* bf_ = (const float*)d_in[17];
  const float* Ws1 = (const float*)d_in[18];
  const float* bs1 = (const float*)d_in[19];
  const float* Ws2 = (const float*)d_in[20];
  const float* bs2 = (const float*)d_in[21];
  float* scores = (float*)d_out;

  // workspace carve-out
  char* wp = (char*)d_ws;
  auto alloc = [&](size_t bytes) -> char* {
    char* p = wp; wp += (bytes + 255) & ~(size_t)255; return p;
  };
  bf16_t* xb    = (bf16_t*)alloc((size_t)N_NODES * F_NODE * sizeof(bf16_t));
  bf16_t* meanb = (bf16_t*)alloc((size_t)N_NODES * H_DIM * sizeof(bf16_t));
  float*  agg   = (float*) alloc((size_t)N_NODES * H_DIM * sizeof(float));
  float*  cnt   = (float*) alloc((size_t)N_NODES * sizeof(float));
  float*  h1f   = (float*) alloc((size_t)N_NODES * H_DIM * sizeof(float));
  bf16_t* h1b   = (bf16_t*)alloc((size_t)N_NODES * H_DIM * sizeof(bf16_t));
  float*  embf  = (float*) alloc((size_t)N_NODES * H_DIM * sizeof(float));
  bf16_t* embb  = (bf16_t*)alloc((size_t)N_NODES * H_DIM * sizeof(bf16_t));
  float*  gsum  = (float*) alloc((size_t)BATCH * H_DIM * sizeof(float));
  float*  gcnt  = (float*) alloc((size_t)BATCH * sizeof(float));
  float*  gemb  = (float*) alloc((size_t)BATCH * H_DIM * sizeof(float));
  float*  femb  = (float*) alloc((size_t)BATCH * 64 * sizeof(float));
  float*  pemb  = (float*) alloc((size_t)BATCH * H_DIM * sizeof(float));
  bf16_t* Wl1b  = (bf16_t*)alloc((size_t)H_DIM * F_NODE * sizeof(bf16_t));
  bf16_t* Wr1b  = (bf16_t*)alloc((size_t)H_DIM * F_NODE * sizeof(bf16_t));
  bf16_t* Wl2b  = (bf16_t*)alloc((size_t)H_DIM * H_DIM * sizeof(bf16_t));
  bf16_t* Wr2b  = (bf16_t*)alloc((size_t)H_DIM * H_DIM * sizeof(bf16_t));
  bf16_t* Wihb  = (bf16_t*)alloc((size_t)4 * H_DIM * H_DIM * sizeof(bf16_t));
  bf16_t* Whhb  = (bf16_t*)alloc((size_t)4 * H_DIM * H_DIM * sizeof(bf16_t));

  auto gsz = [](size_t n) { return (unsigned)((n + 255) / 256); };
  auto zero = [&](float* p, size_t n) {
    zero_f32_kernel<<<gsz(n), 256, 0, stream>>>(p, n);
  };
  auto cvt = [&](const float* s, bf16_t* d, size_t n) {
    f32_to_bf16_kernel<<<gsz(n), 256, 0, stream>>>(s, d, n);
  };

  // ---- precision conversions -------------------------------------------
  cvt(x,   xb,   (size_t)N_NODES * F_NODE);
  cvt(Wl1, Wl1b, (size_t)H_DIM * F_NODE);
  cvt(Wr1, Wr1b, (size_t)H_DIM * F_NODE);
  cvt(Wl2, Wl2b, (size_t)H_DIM * H_DIM);
  cvt(Wr2, Wr2b, (size_t)H_DIM * H_DIM);
  cvt(Wih, Wihb, (size_t)4 * H_DIM * H_DIM);
  cvt(Whh, Whhb, (size_t)4 * H_DIM * H_DIM);

  // ---- SAGE layer 1 -----------------------------------------------------
  zero(agg, (size_t)N_NODES * F_NODE);
  zero(cnt, (size_t)N_NODES);
  edge_count_kernel<<<gsz(N_EDGES), 256, 0, stream>>>(dst, cnt, N_EDGES);
  {
    size_t totq = (size_t)N_EDGES * (F_NODE / 4);
    edge_scatter4_kernel<<<gsz(totq), 256, 0, stream>>>(src, dst, x, agg, F_NODE, totq);
    finalize_mean_kernel<<<gsz((size_t)N_NODES * F_NODE), 256, 0, stream>>>(
        agg, cnt, meanb, F_NODE, (size_t)N_NODES * F_NODE);
  }
  sage_gemm_kernel<<<N_NODES / 128, 256, 0, stream>>>(
      meanb, xb, F_NODE, Wl1b, Wr1b, b1, h1f, h1b);

  // ---- SAGE layer 2 (same degree counts) --------------------------------
  zero(agg, (size_t)N_NODES * H_DIM);
  {
    size_t totq = (size_t)N_EDGES * (H_DIM / 4);
    edge_scatter4_kernel<<<gsz(totq), 256, 0, stream>>>(src, dst, h1f, agg, H_DIM, totq);
    finalize_mean_kernel<<<gsz((size_t)N_NODES * H_DIM), 256, 0, stream>>>(
        agg, cnt, meanb, H_DIM, (size_t)N_NODES * H_DIM);
  }
  sage_gemm_kernel<<<N_NODES / 128, 256, 0, stream>>>(
      meanb, h1b, H_DIM, Wl2b, Wr2b, b2, embf, embb);

  // ---- graph mean pooling ----------------------------------------------
  zero(gsum, (size_t)BATCH * H_DIM);
  zero(gcnt, (size_t)BATCH);
  pool_sum_kernel<<<gsz((size_t)N_NODES * H_DIM), 256, 0, stream>>>(
      embf, batch, gsum, (size_t)N_NODES * H_DIM);
  pool_cnt_kernel<<<gsz(N_NODES), 256, 0, stream>>>(batch, gcnt, N_NODES);
  pool_fin_kernel<<<gsz((size_t)BATCH * H_DIM), 256, 0, stream>>>(gsum, gcnt, gemb);

  // ---- flow embedding ---------------------------------------------------
  flow_kernel<<<BATCH, 64, 0, stream>>>(flow, Wf, bf_, femb);

  // ---- LSTM over paths --------------------------------------------------
  lstm_kernel<<<BATCH / 16, 256, (unsigned)LSTM_SMEM_BYTES, stream>>>(
      embb, path_idx, path_len, Wihb, Whhb, bih, bhh, pemb);

  // ---- scoring head -----------------------------------------------------
  score_kernel<<<BATCH, 128, 0, stream>>>(gemb, pemb, femb, Ws1, bs1, Ws2, bs2, scores);
}
